// MoEAutoEncoder_28363964023540
// MI455X (gfx1250) — matrix operand order repaired and checked by reading, not compiled, not run on hardware
//
#include <hip/hip_runtime.h>
#include <hip/hip_bf16.h>

// ---------------- problem constants ----------------
#define BATCH   4096
#define DIM     768
#define NEXP    16
#define KD      1536
#define TOPE    2
#define KPER    50          // top-k per routed expert

// ---------------- WMMA types ----------------
typedef __attribute__((ext_vector_type(16))) __bf16 v16bf;
typedef __attribute__((ext_vector_type(8)))  float  v8f;

union Frag16 { v16bf v; unsigned int u[8]; };

__device__ __forceinline__ unsigned short f32_to_bf16(float f) {
    unsigned u = __float_as_uint(f);
    unsigned r = u + 0x7fffu + ((u >> 16) & 1u);   // round-to-nearest-even
    return (unsigned short)(r >> 16);
}

// ---------------- workspace layout (bytes, all 256-aligned) ----------------
#define OFF_XB   ((size_t)0)                         // [B][D] bf16            6,291,456
#define OFF_WB   ((size_t)6291456)                   // [E][KD][D] bf16       37,748,736
#define OFF_F    ((size_t)44040192)                  // [B][2][KD] f32        50,331,648
#define OFF_ATOK ((size_t)94371840)                  // [E][B] int               262,144
#define OFF_ASLT ((size_t)94633984)                  // [E][B] int               262,144
#define OFF_AW   ((size_t)94896128)                  // [E][B] f32               262,144
#define OFF_CNT  ((size_t)95158272)                  // [E] int (256B slab)
#define OFF_T2E  ((size_t)95158528)                  // [B][2] int                32,768

// ================= K0a: x - b_dec -> bf16 =================
__global__ void moe_cvt_x(const float* __restrict__ x, const float* __restrict__ b_dec,
                          unsigned short* __restrict__ xb) {
    int i = blockIdx.x * blockDim.x + threadIdx.x;
    if (i < BATCH * DIM) {
        int d = i % DIM;
        xb[i] = f32_to_bf16(x[i] - b_dec[d]);
    }
}

// ================= K0b: Wenc -> bf16 =================
__global__ void moe_cvt_w(const float* __restrict__ w, unsigned short* __restrict__ wb) {
    int i = blockIdx.x * blockDim.x + threadIdx.x;
    if (i < NEXP * KD * DIM) wb[i] = f32_to_bf16(w[i]);
}

// ================= K1: gating + routing (1 wave per token) =================
__global__ void moe_gate(const float* __restrict__ x, const float* __restrict__ Wg,
                         const float* __restrict__ bg, const float* __restrict__ b_gate,
                         int* __restrict__ counts, int* __restrict__ a_tok,
                         int* __restrict__ a_slot, float* __restrict__ a_w,
                         int* __restrict__ t2e) {
    int t = blockIdx.x;
    int lane = threadIdx.x;                 // blockDim == 32 (wave32)
    float acc[NEXP];
    #pragma unroll
    for (int e = 0; e < NEXP; ++e) acc[e] = 0.f;

    for (int d = lane; d < DIM; d += 32) {
        float xv = x[(size_t)t * DIM + d] - b_gate[d];
        #pragma unroll
        for (int e = 0; e < NEXP; ++e) acc[e] += xv * Wg[e * DIM + d];
    }
    #pragma unroll
    for (int e = 0; e < NEXP; ++e)
        for (int off = 16; off > 0; off >>= 1)
            acc[e] += __shfl_xor(acc[e], off, 32);

    if (lane == 0) {
        float p[NEXP], mx = -1e30f;
        #pragma unroll
        for (int e = 0; e < NEXP; ++e) { p[e] = acc[e] + bg[e]; mx = fmaxf(mx, p[e]); }
        float sum = 0.f;
        #pragma unroll
        for (int e = 0; e < NEXP; ++e) { p[e] = __expf(p[e] - mx); sum += p[e]; }
        float inv = 1.f / sum;
        #pragma unroll
        for (int e = 0; e < NEXP; ++e) p[e] *= inv;
        // top-2 (first occurrence wins ties, matching lax.top_k)
        int i1 = 0;
        #pragma unroll
        for (int e = 1; e < NEXP; ++e) if (p[e] > p[i1]) i1 = e;
        int i2 = (i1 == 0) ? 1 : 0;
        #pragma unroll
        for (int e = 0; e < NEXP; ++e) if (e != i1 && p[e] > p[i2]) i2 = e;
        // softmax over the two retained probabilities
        float e1 = __expf(p[i1]), e2 = __expf(p[i2]);
        float w1 = e1 / (e1 + e2), w2 = e2 / (e1 + e2);

        t2e[t * 2 + 0] = i1;
        t2e[t * 2 + 1] = i2;
        int q1 = atomicAdd(&counts[i1], 1);
        a_tok[i1 * BATCH + q1] = t; a_slot[i1 * BATCH + q1] = 0; a_w[i1 * BATCH + q1] = w1;
        int q2 = atomicAdd(&counts[i2], 1);
        a_tok[i2 * BATCH + q2] = t; a_slot[i2 * BATCH + q2] = 1; a_w[i2 * BATCH + q2] = w2;
    }
}

// ================= K2: routed encode GEMM (bf16 WMMA) =================
// grid = (KD/64, 256, NEXP), block = 128 (4 waves); wave w -> one 16x16 col tile.
__global__ void __launch_bounds__(128)
moe_encode(const unsigned short* __restrict__ xb, const unsigned short* __restrict__ wb,
           const int* __restrict__ a_tok, const int* __restrict__ a_slot,
           const float* __restrict__ a_w, const int* __restrict__ counts,
           const float* __restrict__ benc, float* __restrict__ f_buf) {
    __shared__ int   s_tok[16];
    __shared__ int   s_slot[16];
    __shared__ float s_w[16];

    int e   = blockIdx.z;
    int cnt = counts[e];
    int rt  = blockIdx.y;
    if (rt * 16 >= cnt) return;

    int tid = threadIdx.x;
    if (tid < 16) {
        int ai = rt * 16 + tid;
        if (ai < cnt) {
            s_tok[tid]  = a_tok[e * BATCH + ai];
            s_slot[tid] = a_slot[e * BATCH + ai];
            s_w[tid]    = a_w[e * BATCH + ai];
        } else { s_tok[tid] = 0; s_slot[tid] = 0; s_w[tid] = 0.f; }
    }
    __syncthreads();

    int wave = tid >> 5;
    int lane = tid & 31;
    int g    = lane >> 4;          // half-wave id
    int l16  = lane & 15;
    int n    = (blockIdx.x * 4 + wave) * 16 + l16;   // dict column for this lane

    int tokA = s_tok[l16];         // A row m == l16
    const unsigned short* xrow = xb + (size_t)tokA * DIM;
    const unsigned short* wrow = wb + ((size_t)e * KD + n) * DIM;

    v8f acc = {};
    for (int k0 = 0; k0 < DIM; k0 += 32) {
        Frag16 a, b;
        #pragma unroll
        for (int v = 0; v < 8; ++v) {
            int ka = g * 16 + 2 * v;
            int kb = g * 16 + 2 * v;
            a.u[v] = *(const unsigned int*)(xrow + k0 + ka);
            b.u[v] = *(const unsigned int*)(wrow + k0 + kb);
        }
        acc = __builtin_amdgcn_wmma_f32_16x16x32_bf16(false, a.v, false, b.v,
                                                      (short)0, acc, false, false);
    }

    float bia = benc[e * KD + n];
    #pragma unroll
    for (int r = 0; r < 8; ++r) {
        int m  = 8 * g + r;                 // C/D row for this register
        int ai = rt * 16 + m;
        if (ai < cnt) {
            float z = acc[r] + bia;
            z = z > 0.f ? z : 0.f;
            int t = s_tok[m], s = s_slot[m];
            f_buf[((size_t)t * TOPE + s) * KD + n] = z * s_w[m];
        }
    }
}

// ================= K3: exact top-50 per slot + sparse decode =================
__global__ void __launch_bounds__(256)
moe_decode(const float* __restrict__ f_buf, const int* __restrict__ t2e,
           const float* __restrict__ Wdec, const float* __restrict__ b_dec,
           float* __restrict__ xhat) {
    __shared__ float s_f[TOPE * KD];
    __shared__ unsigned char s_flag[TOPE * KD];
    __shared__ int   s_scan[256];
    __shared__ int   s_sel_idx[128];
    __shared__ float s_sel_val[128];
    __shared__ int   s_red;
    __shared__ int   s_cg[2];

    int t   = blockIdx.x;
    int tid = threadIdx.x;

    for (int i = tid; i < TOPE * KD; i += 256)
        s_f[i] = f_buf[(size_t)t * TOPE * KD + i];
    __syncthreads();

    // --- exact 50th-largest value per slot via bitwise search (vals >= 0) ---
    unsigned v50[2];
    for (int s = 0; s < 2; ++s) {
        unsigned ans = 0;
        for (int bit = 30; bit >= 0; --bit) {
            unsigned cand = ans | (1u << bit);
            int c = 0;
            for (int i = tid; i < KD; i += 256)
                c += (__float_as_uint(s_f[s * KD + i]) >= cand);
            __syncthreads();
            if (tid == 0) s_red = 0;
            __syncthreads();
            atomicAdd(&s_red, c);
            __syncthreads();
            if (s_red >= KPER) ans = cand;
        }
        v50[s] = ans;
    }

    // --- count strictly-greater per slot ---
    int c0 = 0, c1 = 0;
    for (int i = tid; i < KD; i += 256) {
        c0 += (__float_as_uint(s_f[i])      > v50[0]);
        c1 += (__float_as_uint(s_f[KD + i]) > v50[1]);
    }
    __syncthreads();
    if (tid == 0) { s_cg[0] = 0; s_cg[1] = 0; }
    __syncthreads();
    atomicAdd(&s_cg[0], c0);
    atomicAdd(&s_cg[1], c1);
    __syncthreads();

    // --- selection flags ---
    for (int i = tid; i < TOPE * KD; i += 256) {
        unsigned thr = v50[i >= KD];
        s_flag[i] = (__float_as_uint(s_f[i]) > thr) ? 1 : 0;
    }
    __syncthreads();
    if (tid == 0) {  // fill ties at threshold in index order (deterministic)
        for (int s = 0; s < 2; ++s) {
            if (v50[s] != 0u) {
                int need = KPER - s_cg[s];
                for (int i = 0; i < KD && need > 0; ++i) {
                    if (__float_as_uint(s_f[s * KD + i]) == v50[s]) { s_flag[s * KD + i] = 1; --need; }
                }
            }
        }
    }
    __syncthreads();

    // --- stable compaction: 12 contiguous entries per thread + block scan ---
    int base = tid * 12, cc = 0;
    #pragma unroll
    for (int j = 0; j < 12; ++j) cc += s_flag[base + j];
    s_scan[tid] = cc;
    __syncthreads();
    for (int off = 1; off < 256; off <<= 1) {
        int vv = s_scan[tid];
        int ad = (tid >= off) ? s_scan[tid - off] : 0;
        __syncthreads();
        s_scan[tid] = vv + ad;
        __syncthreads();
    }
    int k = s_scan[tid] - cc;
    #pragma unroll
    for (int j = 0; j < 12; ++j) {
        int gi = base + j;
        if (s_flag[gi] && k < 128) { s_sel_idx[k] = gi; s_sel_val[k] = s_f[gi]; ++k; }
    }
    __syncthreads();
    int total = s_scan[255];
    if (total > 128) total = 128;

    // --- accumulate x_hat: each thread owns dims tid, tid+256, tid+512 ---
    int e0 = t2e[t * 2 + 0], e1 = t2e[t * 2 + 1];
    float a0 = 0.f, a1 = 0.f, a2 = 0.f;
    for (int i = 0; i < total; ++i) {
        int gi = s_sel_idx[i];
        float val = s_sel_val[i];
        int s   = (gi >= KD);
        int idx = gi - s * KD;
        int eid = s ? e1 : e0;
        const float* wr = Wdec + ((size_t)eid * KD + idx) * DIM;
        a0 += val * wr[tid];
        a1 += val * wr[tid + 256];
        a2 += val * wr[tid + 512];
    }
    xhat[(size_t)t * DIM + tid]       = a0 + b_dec[tid];
    xhat[(size_t)t * DIM + tid + 256] = a1 + b_dec[tid + 256];
    xhat[(size_t)t * DIM + tid + 512] = a2 + b_dec[tid + 512];
}

// ================= launch =================
extern "C" void kernel_launch(void* const* d_in, const int* in_sizes, int n_in,
                              void* d_out, int out_size, void* d_ws, size_t ws_size,
                              hipStream_t stream) {
    const float* x      = (const float*)d_in[0];
    const float* Wg     = (const float*)d_in[1];
    const float* bg     = (const float*)d_in[2];
    const float* b_gate = (const float*)d_in[3];
    const float* b_dec  = (const float*)d_in[4];
    const float* Wenc   = (const float*)d_in[5];
    const float* benc   = (const float*)d_in[6];
    const float* Wdec   = (const float*)d_in[7];
    float* xhat = (float*)d_out;

    char* ws = (char*)d_ws;
    unsigned short* xb   = (unsigned short*)(ws + OFF_XB);
    unsigned short* wb   = (unsigned short*)(ws + OFF_WB);
    float*          fbuf = (float*)(ws + OFF_F);
    int*            atok = (int*)(ws + OFF_ATOK);
    int*            aslt = (int*)(ws + OFF_ASLT);
    float*          aw   = (float*)(ws + OFF_AW);
    int*            cnt  = (int*)(ws + OFF_CNT);
    int*            t2e  = (int*)(ws + OFF_T2E);

    hipMemsetAsync(cnt, 0, 256, stream);

    moe_cvt_x<<<(BATCH * DIM + 255) / 256, 256, 0, stream>>>(x, b_dec, xb);
    moe_cvt_w<<<(NEXP * KD * DIM + 255) / 256, 256, 0, stream>>>(Wenc, wb);
    moe_gate<<<BATCH, 32, 0, stream>>>(x, Wg, bg, b_gate, cnt, atok, aslt, aw, t2e);
    moe_encode<<<dim3(KD / 64, BATCH / 16, NEXP), 128, 0, stream>>>(
        xb, wb, atok, aslt, aw, cnt, benc, fbuf);
    moe_decode<<<BATCH, 256, 0, stream>>>(fbuf, t2e, Wdec, b_dec, xhat);
}
